// GPTLanguageModel_89627377533678
// MI455X (gfx1250) — matrix operand order repaired and checked by reading, not compiled, not run on hardware
//
#include <hip/hip_runtime.h>

// ---------------------------------------------------------------------------
// GPT forward for gfx1250 (MI455X, wave32, WMMA + Tensor Data Mover).
// Weights pre-converted once per launch to bf16 transposed [N][K]; all
// activations carried in bf16 for GEMM/WMMA, f32 kept only for the residual
// stream / LayerNorm math. GEMM tiles double-buffered in LDS and streamed by
// TDM (tensor_load_to_lds + s_wait_tensorcnt) overlapping DMA with WMMA.
// ---------------------------------------------------------------------------

typedef __attribute__((ext_vector_type(16))) __bf16       v16bf;
typedef __attribute__((ext_vector_type(8)))  float        v8f;
typedef __attribute__((ext_vector_type(8)))  unsigned int v8u;
typedef __attribute__((ext_vector_type(4)))  unsigned int v4u;
typedef __attribute__((ext_vector_type(8)))  int          v8i;
typedef __attribute__((ext_vector_type(4)))  int          v4i;

static constexpr int kB  = 2;
static constexpr int kT  = 2048;
static constexpr int kE  = 1024;
static constexpr int kH  = 16;
static constexpr int kDH = 64;
static constexpr int kV  = 32000;
static constexpr int kL  = 8;
static constexpr int kBT = kB * kT;  // 4096

#if defined(__AMDGCN__) && __has_builtin(__builtin_amdgcn_tensor_load_to_lds) && \
    __has_builtin(__builtin_amdgcn_s_wait_tensorcnt)
#define USE_TDM 1
#if __clang_major__ >= 23
#define TDM_SIX 1
#endif
#endif

// ---- helpers ---------------------------------------------------------------

__device__ __forceinline__ unsigned short f2bf(float f) {
  unsigned u = __float_as_uint(f);
  u += 0x7fffu + ((u >> 16) & 1u);   // round-to-nearest-even
  return (unsigned short)(u >> 16);
}
__device__ __forceinline__ unsigned pack2bf(float a, float b) {
  return (unsigned)f2bf(a) | ((unsigned)f2bf(b) << 16);
}
__device__ __forceinline__ v8f wmma_bf16(v8u a, v8u b, v8f c) {
  union U { v8u u; v16bf h; };
  U ua; ua.u = a;
  U ub; ub.u = b;
  return __builtin_amdgcn_wmma_f32_16x16x32_bf16(false, ua.h, false, ub.h,
                                                 (short)0, c, false, false);
}

#if USE_TDM
// 2D tile load Global->LDS via Tensor Data Mover (ISA cdna5 ch.7/8).
// D# group0: count=1, lds_addr, global_addr, type=2.
// D# group1: data_size=2B, pad_enable + pad_interval/pad_amount (LDS row
// padding), tensor_dim0/1, tile_dim0/1, tensor_dim0_stride (elements).
__device__ __forceinline__ void tdm_load_2d(
    unsigned lds_addr, const void* gptr,
    unsigned tile0, unsigned tile1, unsigned stride0,
    unsigned pad_int, unsigned pad_amt)
{
  const unsigned long long ga = (unsigned long long)(size_t)gptr;
  v4u g0;
  g0[0] = 1u;                                   // count=1 (valid descriptor)
  g0[1] = lds_addr;                             // LDS byte address
  g0[2] = (unsigned)ga;                         // global_addr[31:0]
  g0[3] = (unsigned)(ga >> 32) | 0x80000000u;   // global_addr[56:32] | type=2
  v8i g1;
  g1[0] = (int)((1u << 16) | (1u << 20) | (pad_int << 22) | (pad_amt << 25));
  g1[1] = (int)((tile0 & 0xFFFFu) << 16);       // tensor_dim0 = tile0 (no OOB)
  g1[2] = (int)((tile0 >> 16) | ((tile1 & 0xFFFFu) << 16)); // tensor_dim1
  g1[3] = (int)((tile1 >> 16) | (tile0 << 16)); // tile_dim0
  g1[4] = (int)(tile1 & 0xFFFFu);               // tile_dim1 (tile_dim2=0)
  g1[5] = (int)stride0;                         // tensor_dim0_stride[31:0]
  g1[6] = 0;
  g1[7] = 0;
  v4i z4; z4[0] = z4[1] = z4[2] = z4[3] = 0;    // groups 2/3 unused (2D)
#ifdef TDM_SIX
  v8i z8;
  for (int i = 0; i < 8; ++i) z8[i] = 0;
  __builtin_amdgcn_tensor_load_to_lds(g0, g1, z4, z4, z8, 0);
#else
  __builtin_amdgcn_tensor_load_to_lds(g0, g1, z4, z4, 0);
#endif
}
#endif

// ---- weight convert+transpose: W f32 [K][N] -> Wt bf16 [N][K] --------------

__global__ __launch_bounds__(256) void wconv_kernel(
    const float* __restrict__ W, unsigned short* __restrict__ Wt, int K, int N)
{
  __shared__ float tile[32][33];
  const int kb = blockIdx.y * 32;
  const int nb = blockIdx.x * 32;
  const size_t mat = (size_t)blockIdx.z * K * N;
  const int tx = threadIdx.x & 31;
  const int ty = threadIdx.x >> 5;  // 0..7
#pragma unroll
  for (int i = 0; i < 4; ++i)
    tile[ty + 8 * i][tx] = W[mat + (size_t)(kb + ty + 8 * i) * N + (nb + tx)];
  __syncthreads();
#pragma unroll
  for (int i = 0; i < 4; ++i)
    Wt[mat + (size_t)(nb + ty + 8 * i) * K + (kb + tx)] =
        f2bf(tile[tx][ty + 8 * i]);
}

// ---- GEMM: C[M,N] = A[M,K](bf16) * Bt[N,K](bf16,transposed) ----------------
// Block 128x128x32, 256 threads = 8 waves (2x4), wave = 64x32 (4x2 WMMA).
// Double-buffered LDS tiles streamed by TDM: wave0 issues the next tile pair,
// then s_wait_tensorcnt 2 (TDM in-order per wave => current pair landed)
// while the in-flight pair overlaps with WMMA on the current buffer.

__global__ __launch_bounds__(256) void gemm_kernel(
    const unsigned short* __restrict__ A, const unsigned short* __restrict__ Bt,
    const float* __restrict__ bias, float* __restrict__ Cf,
    unsigned short* __restrict__ Cb, int M, int N, int K, int relu)
{
  __shared__ __align__(16) unsigned short a_sh[2][128][34];  // [buf][m][k]
  __shared__ __align__(16) unsigned short b_sh[2][128][34];  // [buf][n][k]

  const int tid  = threadIdx.x;
  const int lane = tid & 31;
  const int wv   = tid >> 5;
  const int wm   = wv >> 2;
  const int wn   = wv & 3;
  const int l15  = lane & 15;
  const int half = lane >> 4;
  const int bm   = blockIdx.y * 128;
  const int bn   = blockIdx.x * 128;
  (void)M;

  v8f acc[4][2];
#pragma unroll
  for (int i = 0; i < 4; ++i)
#pragma unroll
    for (int j = 0; j < 2; ++j)
#pragma unroll
      for (int e = 0; e < 8; ++e) acc[i][j][e] = 0.0f;

  const int nsteps = K >> 5;

#if USE_TDM
  const unsigned a_lds[2] = {(unsigned)(size_t)&a_sh[0][0][0],
                             (unsigned)(size_t)&a_sh[1][0][0]};
  const unsigned b_lds[2] = {(unsigned)(size_t)&b_sh[0][0][0],
                             (unsigned)(size_t)&b_sh[1][0][0]};
  if (wv == 0) {  // prologue: stream tile 0 into buffer 0
    tdm_load_2d(a_lds[0], A + (size_t)bm * K, 32, 128, (unsigned)K, 3, 0);
    tdm_load_2d(b_lds[0], Bt + (size_t)bn * K, 32, 128, (unsigned)K, 3, 0);
  }
#endif

  for (int s = 0; s < nsteps; ++s) {
    const int cur = s & 1;
    __syncthreads();  // all waves done consuming buffer cur^1
#if USE_TDM
    if (wv == 0) {
      if (s + 1 < nsteps) {
        const int k1 = (s + 1) << 5;
        tdm_load_2d(a_lds[cur ^ 1], A + (size_t)bm * K + k1, 32, 128,
                    (unsigned)K, 3, 0);
        tdm_load_2d(b_lds[cur ^ 1], Bt + (size_t)bn * K + k1, 32, 128,
                    (unsigned)K, 3, 0);
        __builtin_amdgcn_s_wait_tensorcnt(2);  // tile s landed; s+1 in flight
      } else {
        __builtin_amdgcn_s_wait_tensorcnt(0);
      }
    }
#else
    {
      const int cr = tid >> 1;
      const int cs = (tid & 1) * 16;
      const int k0 = s << 5;
      const unsigned* asrc =
          (const unsigned*)(A + (size_t)(bm + cr) * K + k0 + cs);
      unsigned* adst = (unsigned*)&a_sh[cur][cr][cs];
#pragma unroll
      for (int i = 0; i < 8; ++i) adst[i] = asrc[i];
      const unsigned* bsrc =
          (const unsigned*)(Bt + (size_t)(bn + cr) * K + k0 + cs);
      unsigned* bdst = (unsigned*)&b_sh[cur][cr][cs];
#pragma unroll
      for (int i = 0; i < 8; ++i) bdst[i] = bsrc[i];
    }
#endif
    __syncthreads();  // buffer cur ready

    // B fragments: lane: n = tileN + l15, K = half*16 + 0..15
    v8u bfrag[2];
#pragma unroll
    for (int fn = 0; fn < 2; ++fn) {
      const unsigned* src =
          (const unsigned*)&b_sh[cur][wn * 32 + fn * 16 + l15][half * 16];
#pragma unroll
      for (int i = 0; i < 8; ++i) bfrag[fn][i] = src[i];
    }
    // A fragments: lane: row = tileM + l15, K = half*16 + 0..15
#pragma unroll
    for (int fm = 0; fm < 4; ++fm) {
      const unsigned* src =
          (const unsigned*)&a_sh[cur][wm * 64 + fm * 16 + l15][0];
      v8u afrag;
      const int kq = half * 8;
#pragma unroll
      for (int i = 0; i < 4; ++i) {
        afrag[i]     = src[kq + i];
        afrag[4 + i] = src[kq + 4 + i];
      }
#pragma unroll
      for (int fn = 0; fn < 2; ++fn)
        acc[fm][fn] = wmma_bf16(afrag, bfrag[fn], acc[fm][fn]);
    }
  }

#pragma unroll
  for (int fm = 0; fm < 4; ++fm)
#pragma unroll
    for (int fn = 0; fn < 2; ++fn) {
      const int col = bn + wn * 32 + fn * 16 + l15;
      const float bv = bias ? bias[col] : 0.0f;
#pragma unroll
      for (int j = 0; j < 8; ++j) {
        const int row = bm + wm * 64 + fm * 16 + j + 8 * half;
        float r = acc[fm][fn][j] + bv;
        if (relu) r = fmaxf(r, 0.0f);
        if (Cf) Cf[(size_t)row * N + col] = r;
        if (Cb) Cb[(size_t)row * N + col] = f2bf(r);
      }
    }
}

// ---- Flash-style causal attention (bf16 q/k/v in, bf16 out) ----------------
// grid = (T/64, B*H), block = 128 (4 waves), 16 query rows per wave.

__global__ __launch_bounds__(128) void attn_kernel(
    const unsigned short* __restrict__ Q, const unsigned short* __restrict__ Kb,
    const unsigned short* __restrict__ Vb, unsigned short* __restrict__ O)
{
  __shared__ __align__(16) unsigned short k_sh[32][68];  // [key][dh]
  __shared__ __align__(16) unsigned short v_sh[64][36];  // [dh][key]
  __shared__ float p_sh[4][16][32];

  const int tid  = threadIdx.x;
  const int lane = tid & 31;
  const int wv   = tid >> 5;
  const int l15  = lane & 15;
  const int half = lane >> 4;

  const int qblk  = blockIdx.x * 64;
  const int bh    = blockIdx.y;
  const int b     = bh >> 4;   // H = 16
  const int h     = bh & 15;
  const int qbase = qblk + wv * 16;

  // Q fragments: 16 rows x 64 dh -> two 16x32 A fragments (direct bf16 loads)
  v8u qfrag[2];
  {
    const int trow = qbase + l15;
    const unsigned* q32 =
        (const unsigned*)(Q + ((size_t)(b * kT + trow)) * kE + h * kDH);
#pragma unroll
    for (int hh = 0; hh < 2; ++hh) {
      const int dw = hh * 16 + half * 8;
#pragma unroll
      for (int i = 0; i < 4; ++i) {
        qfrag[hh][i]     = q32[dw + i];
        qfrag[hh][4 + i] = q32[dw + 4 + i];
      }
    }
  }

  v8f ofrag[4];
#pragma unroll
  for (int fn = 0; fn < 4; ++fn)
#pragma unroll
    for (int j = 0; j < 8; ++j) ofrag[fn][j] = 0.0f;

  float mrow[8], lrow[8];
#pragma unroll
  for (int j = 0; j < 8; ++j) { mrow[j] = -1e30f; lrow[j] = 0.0f; }

  const float scale = 0.125f;
  const int kend = qblk + 64;

#if USE_TDM
  const unsigned k_lds = (unsigned)(size_t)&k_sh[0][0];
#endif

  for (int kb = 0; kb < kend; kb += 32) {
    __syncthreads();
#if USE_TDM
    if (wv == 0)   // K tile: 32 keys x 64 dh, +2-dword row pad -> stride 68
      tdm_load_2d(k_lds, Kb + ((size_t)(b * kT + kb)) * kE + h * kDH,
                  64, 32, (unsigned)kE, 4, 1);
#else
    {
      const int key = tid >> 2;
      const int seg = (tid & 3) * 16;
      const unsigned* ksrc =
          (const unsigned*)(Kb + ((size_t)(b * kT + kb + key)) * kE + h * kDH + seg);
      unsigned* kdst = (unsigned*)&k_sh[key][seg];
#pragma unroll
      for (int i = 0; i < 8; ++i) kdst[i] = ksrc[i];
    }
#endif
    {  // V tile transposed (scatter; TDM cannot transpose) — overlaps TDM
      const int key = tid >> 2;
      const int seg = (tid & 3) * 16;
      const unsigned short* vp =
          Vb + ((size_t)(b * kT + kb + key)) * kE + h * kDH + seg;
#pragma unroll
      for (int i = 0; i < 16; ++i) v_sh[seg + i][key] = vp[i];
    }
#if USE_TDM
    if (wv == 0) __builtin_amdgcn_s_wait_tensorcnt(0);
#endif
    __syncthreads();

    // scores s(16q x 32keys): inner dim DH=64 = 2 WMMA k-steps
    v8f sfr[2];
#pragma unroll
    for (int c = 0; c < 2; ++c) {
#pragma unroll
      for (int e = 0; e < 8; ++e) sfr[c][e] = 0.0f;
      const int key = c * 16 + l15;
#pragma unroll
      for (int hh = 0; hh < 2; ++hh) {
        const unsigned* src = (const unsigned*)&k_sh[key][hh * 32 + half * 16];
        v8u kf;
#pragma unroll
        for (int i = 0; i < 8; ++i) kf[i] = src[i];
        sfr[c] = wmma_bf16(qfrag[hh], kf, sfr[c]);
      }
    }

    // online softmax (row = j + 8*half, striped across 16-lane halves)
    float alph[8];
#pragma unroll
    for (int j = 0; j < 8; ++j) {
      const int qidx = qbase + j + 8 * half;
      float s0 = sfr[0][j] * scale;
      float s1 = sfr[1][j] * scale;
      if (kb + l15 > qidx)      s0 = -1e30f;
      if (kb + 16 + l15 > qidx) s1 = -1e30f;
      float mx = fmaxf(s0, s1);
      mx = fmaxf(mx, __shfl_xor(mx, 1, 32));
      mx = fmaxf(mx, __shfl_xor(mx, 2, 32));
      mx = fmaxf(mx, __shfl_xor(mx, 4, 32));
      mx = fmaxf(mx, __shfl_xor(mx, 8, 32));
      const float mnew = fmaxf(mrow[j], mx);
      const float p0 = __expf(s0 - mnew);
      const float p1 = __expf(s1 - mnew);
      float rs = p0 + p1;
      rs += __shfl_xor(rs, 1, 32);
      rs += __shfl_xor(rs, 2, 32);
      rs += __shfl_xor(rs, 4, 32);
      rs += __shfl_xor(rs, 8, 32);
      const float a = __expf(mrow[j] - mnew);
      lrow[j] = lrow[j] * a + rs;
      mrow[j] = mnew;
      alph[j] = a;
      const int mloc = j + 8 * half;
      p_sh[wv][mloc][l15]      = p0;
      p_sh[wv][mloc][16 + l15] = p1;
    }
#pragma unroll
    for (int fn = 0; fn < 4; ++fn)
#pragma unroll
      for (int j = 0; j < 8; ++j) ofrag[fn][j] *= alph[j];

    __syncthreads();

    // P C-layout -> A-layout (bf16), then PV WMMAs over 4 dh tiles
    v8u pfr;
    {
      const float* pr = &p_sh[wv][l15][0];
      const int k0 = half * 16;
#pragma unroll
      for (int i = 0; i < 4; ++i) {
        pfr[i]     = pack2bf(pr[k0 + 2 * i],      pr[k0 + 2 * i + 1]);
        pfr[4 + i] = pack2bf(pr[k0 + 8 + 2 * i], pr[k0 + 8 + 2 * i + 1]);
      }
    }
#pragma unroll
    for (int fn = 0; fn < 4; ++fn) {
      const unsigned* src = (const unsigned*)&v_sh[fn * 16 + l15][half * 16];
      v8u vf;
#pragma unroll
      for (int i = 0; i < 8; ++i) vf[i] = src[i];
      ofrag[fn] = wmma_bf16(pfr, vf, ofrag[fn]);
    }
  }

#pragma unroll
  for (int fn = 0; fn < 4; ++fn)
#pragma unroll
    for (int j = 0; j < 8; ++j) {
      const int trow = qbase + j + 8 * half;
      const int dh = fn * 16 + l15;
      O[((size_t)(b * kT + trow)) * kE + h * kDH + dh] =
          f2bf(ofrag[fn][j] / lrow[j]);
    }
}

// ---- embedding: x (f32) and xb (bf16) --------------------------------------

__global__ __launch_bounds__(256) void embed_kernel(
    const int* __restrict__ idx, const float* __restrict__ tok,
    const float* __restrict__ pos, float* __restrict__ X,
    unsigned short* __restrict__ Xb)
{
  const int bt = blockIdx.x;
  const int t = bt & (kT - 1);
  const int id = idx[bt];
#pragma unroll
  for (int i = 0; i < 4; ++i) {
    const int e = threadIdx.x + 256 * i;
    const float v = tok[(size_t)id * kE + e] + pos[(size_t)t * kE + e];
    X[(size_t)bt * kE + e] = v;
    Xb[(size_t)bt * kE + e] = f2bf(v);
  }
}

// ---- residual add + LayerNorm (Y may be null); writes f32 + bf16 -----------

__global__ __launch_bounds__(256) void ln_kernel(
    const float* __restrict__ X, const float* __restrict__ Y,
    const float* __restrict__ s, const float* __restrict__ bsh,
    float* __restrict__ Out, unsigned short* __restrict__ Outb)
{
  __shared__ float red[256];
  const int row = blockIdx.x;
  const int tid = threadIdx.x;
  float vals[4];
  float lsum = 0.0f;
#pragma unroll
  for (int i = 0; i < 4; ++i) {
    const int e = tid + 256 * i;
    float t = X[(size_t)row * kE + e];
    if (Y) t += Y[(size_t)row * kE + e];
    vals[i] = t;
    lsum += t;
  }
  red[tid] = lsum;
  __syncthreads();
  for (int off = 128; off > 0; off >>= 1) {
    if (tid < off) red[tid] += red[tid + off];
    __syncthreads();
  }
  const float mean = red[0] * (1.0f / kE);
  __syncthreads();
  float lsq = 0.0f;
#pragma unroll
  for (int i = 0; i < 4; ++i) { const float d = vals[i] - mean; lsq += d * d; }
  red[tid] = lsq;
  __syncthreads();
  for (int off = 128; off > 0; off >>= 1) {
    if (tid < off) red[tid] += red[tid + off];
    __syncthreads();
  }
  const float rstd = rsqrtf(red[0] * (1.0f / kE) + 1e-5f);
#pragma unroll
  for (int i = 0; i < 4; ++i) {
    const int e = tid + 256 * i;
    const float r = (vals[i] - mean) * rstd * s[e] + bsh[e];
    Out[(size_t)row * kE + e] = r;
    Outb[(size_t)row * kE + e] = f2bf(r);
  }
}

// ---- host orchestration ----------------------------------------------------

extern "C" void kernel_launch(void* const* d_in, const int* in_sizes, int n_in,
                              void* d_out, int out_size, void* d_ws, size_t ws_size,
                              hipStream_t stream)
{
  const int*   index   = (const int*)  d_in[0];
  const float* tok_emb = (const float*)d_in[1];
  const float* pos_emb = (const float*)d_in[2];
  const float* wq      = (const float*)d_in[3];
  const float* wk      = (const float*)d_in[4];
  const float* wvw     = (const float*)d_in[5];
  const float* wproj   = (const float*)d_in[6];
  const float* bproj   = (const float*)d_in[7];
  const float* w1      = (const float*)d_in[8];
  const float* b1      = (const float*)d_in[9];
  const float* w2      = (const float*)d_in[10];
  const float* b2      = (const float*)d_in[11];
  const float* ln1s    = (const float*)d_in[12];
  const float* ln1b    = (const float*)d_in[13];
  const float* ln2s    = (const float*)d_in[14];
  const float* ln2b    = (const float*)d_in[15];
  const float* lnfs    = (const float*)d_in[16];
  const float* lnfb    = (const float*)d_in[17];
  const float* wlm     = (const float*)d_in[18];
  const float* blm     = (const float*)d_in[19];

  const size_t act = (size_t)kBT * kE;
  char* p = (char*)d_ws;
  float* xf = (float*)p;          p += act * 4;
  float* y  = (float*)p;          p += act * 4;
  unsigned short* xb  = (unsigned short*)p; p += act * 2;
  unsigned short* qb  = (unsigned short*)p; p += act * 2;
  unsigned short* kbu = (unsigned short*)p; p += act * 2;
  unsigned short* vbu = (unsigned short*)p; p += act * 2;
  unsigned short* aob = (unsigned short*)p; p += act * 2;
  unsigned short* hb  = (unsigned short*)p; p += (size_t)kBT * 4 * kE * 2;
  unsigned short* wtq = (unsigned short*)p; p += (size_t)kL * kE * kE * 2;
  unsigned short* wtk = (unsigned short*)p; p += (size_t)kL * kE * kE * 2;
  unsigned short* wtv = (unsigned short*)p; p += (size_t)kL * kE * kE * 2;
  unsigned short* wtp = (unsigned short*)p; p += (size_t)kL * kE * kE * 2;
  unsigned short* wt1 = (unsigned short*)p; p += (size_t)kL * kE * 4 * kE * 2;
  unsigned short* wt2 = (unsigned short*)p; p += (size_t)kL * kE * 4 * kE * 2;
  unsigned short* wtl = (unsigned short*)p;

  // weight convert + transpose to bf16 [N][K]
  wconv_kernel<<<dim3(kE / 32, kE / 32, kL), 256, 0, stream>>>(wq, wtq, kE, kE);
  wconv_kernel<<<dim3(kE / 32, kE / 32, kL), 256, 0, stream>>>(wk, wtk, kE, kE);
  wconv_kernel<<<dim3(kE / 32, kE / 32, kL), 256, 0, stream>>>(wvw, wtv, kE, kE);
  wconv_kernel<<<dim3(kE / 32, kE / 32, kL), 256, 0, stream>>>(wproj, wtp, kE, kE);
  wconv_kernel<<<dim3(4 * kE / 32, kE / 32, kL), 256, 0, stream>>>(w1, wt1, kE, 4 * kE);
  wconv_kernel<<<dim3(kE / 32, 4 * kE / 32, kL), 256, 0, stream>>>(w2, wt2, 4 * kE, kE);
  wconv_kernel<<<dim3(kV / 32, kE / 32, 1), 256, 0, stream>>>(wlm, wtl, kE, kV);

  embed_kernel<<<kBT, 256, 0, stream>>>(index, tok_emb, pos_emb, xf, xb);

  const dim3 gE(kE / 128, kBT / 128);        // 8 x 32
  const dim3 gF(4 * kE / 128, kBT / 128);    // 32 x 32
  const dim3 gA(kT / 64, kB * kH);           // 32 x 32
  const dim3 gV(kV / 128, kBT / 128);        // 250 x 32

  for (int l = 0; l < kL; ++l) {
    const size_t wo  = (size_t)l * kE * kE;
    const size_t w1o = (size_t)l * kE * 4 * kE;
    const size_t eo  = (size_t)l * kE;
    const size_t fo  = (size_t)l * 4 * kE;

    gemm_kernel<<<gE, 256, 0, stream>>>(xb, wtq + wo, nullptr, nullptr, qb,
                                        kBT, kE, kE, 0);
    gemm_kernel<<<gE, 256, 0, stream>>>(xb, wtk + wo, nullptr, nullptr, kbu,
                                        kBT, kE, kE, 0);
    gemm_kernel<<<gE, 256, 0, stream>>>(xb, wtv + wo, nullptr, nullptr, vbu,
                                        kBT, kE, kE, 0);
    attn_kernel<<<gA, 128, 0, stream>>>(qb, kbu, vbu, aob);
    gemm_kernel<<<gE, 256, 0, stream>>>(aob, wtp + wo, bproj + eo, y, nullptr,
                                        kBT, kE, kE, 0);
    ln_kernel<<<kBT, 256, 0, stream>>>(xf, y, ln1s + eo, ln1b + eo, xf, xb);
    gemm_kernel<<<gF, 256, 0, stream>>>(xb, wt1 + w1o, b1 + fo, nullptr, hb,
                                        kBT, 4 * kE, kE, 1);
    gemm_kernel<<<gE, 256, 0, stream>>>(hb, wt2 + w1o, b2 + eo, y, nullptr,
                                        kBT, kE, 4 * kE, 0);
    ln_kernel<<<kBT, 256, 0, stream>>>(xf, y, ln2s + eo, ln2b + eo, xf, xb);
  }
  ln_kernel<<<kBT, 256, 0, stream>>>(xf, nullptr, lnfs, lnfb, xf, xb);
  gemm_kernel<<<gV, 256, 0, stream>>>(xb, wtl, blm, (float*)d_out, nullptr,
                                      kBT, kV, kE, 0);
}
